// xPaiNN_3066606649503
// MI455X (gfx1250) — matrix operand order repaired and checked by reading, not compiled, not run on hardware
//
#include <hip/hip_runtime.h>
#include <math.h>
#include <stdint.h>

#define N_NODES 50000
#define N_EDGES 600000
#define DIM     128
#define D3      384
#define NB      20
#define NLAYER  3
#define HD      64
#define NGRAPH  512
#define CUTOFF  5.0f
#define PI_F    3.14159265358979f
#define EPB     16   // edges per block in message kernel

typedef __attribute__((ext_vector_type(16))) __bf16 v16bf;
typedef __attribute__((ext_vector_type(8)))  __bf16 v8bf;
typedef __attribute__((ext_vector_type(4)))  __bf16 v4bf;
typedef __attribute__((ext_vector_type(8)))  float  v8f;
typedef __attribute__((ext_vector_type(4)))  float  f4;

// fast silu: x * v_rcp_f32(1+exp(-x))  (avoids IEEE div_scale/div_fixup chain)
__device__ inline float silu_f(float x) {
    return x * __builtin_amdgcn_rcpf(1.0f + __expf(-x));
}

__device__ inline void atomic_add_f32(float* p, float v) {
    __hip_atomic_fetch_add(p, v, __ATOMIC_RELAXED, __HIP_MEMORY_SCOPE_AGENT);
}

// ---- CDNA5 async global->LDS copy (ASYNCcnt-tracked), CDNA5 ISA ----
__device__ inline void async_copy_b128_to_lds(uint32_t lds_off, const void* gptr) {
    asm volatile("global_load_async_to_lds_b128 %0, %1, off"
                 :: "v"(lds_off), "v"((uint64_t)(uintptr_t)gptr)
                 : "memory");
}
__device__ inline void wait_asynccnt0() {
    asm volatile("s_wait_asynccnt 0x0" ::: "memory");
}

// ---- WMMA fragment helpers (CDNA5 wave32 layouts, CDNA5 ISA §7.12.2) ----

// A: 16x32 bf16 tile at column kb*32 of a row-major [16 x lda] bf16 LDS buffer.
// Per lane this is two contiguous 16B runs -> two ds_load_b128.
__device__ inline v16bf a_frag(const __bf16* A, int lda, int kb, int lane) {
    int m = lane & 15, half = lane >> 4;
    const __bf16* r = A + m * lda + kb * 32 + half * 8;
    v8bf lo = *(const v8bf*)r;
    v8bf hi = *(const v8bf*)(r + 16);
    return __builtin_shufflevector(lo, hi, 0, 1, 2, 3, 4, 5, 6, 7,
                                           8, 9, 10, 11, 12, 13, 14, 15);
}

// B: fragment-packed weights -- lane's 16 bf16 contiguous (32B) -> two global_load_b128.
__device__ inline v16bf b_frag_packed(const __bf16* Wp, int nkb, int kb, int nt, int lane) {
    const __bf16* r = Wp + (((size_t)(nt * nkb + kb) * 32 + lane) << 4);
    v8bf lo = *(const v8bf*)r;
    v8bf hi = *(const v8bf*)(r + 8);
    return __builtin_shufflevector(lo, hi, 0, 1, 2, 3, 4, 5, 6, 7,
                                           8, 9, 10, 11, 12, 13, 14, 15);
}

#define WMMA_BF16(a, b, c) \
    __builtin_amdgcn_wmma_f32_16x16x32_bf16(false, (a), false, (b), (short)0, (c), false, false)

// ---------------------------------------------------------------------------
// Weight repack: f32 [K x C] row-major -> bf16 WMMA-B fragment order.
// packed[((nt*nkb + kb)*32 + lane)*16 + j] = W[(kb*32 + (j&8)*2 + (lane>>4)*8 + (j&7))*C + nt*16 + (lane&15)]
// ---------------------------------------------------------------------------
__global__ void pack_b_kernel(const float* __restrict__ W, __bf16* __restrict__ Wp,
                              int K, int C) {
    int i = blockIdx.x * blockDim.x + threadIdx.x;
    if (i >= K * C) return;
    int j    = i & 15;
    int lane = (i >> 4) & 31;
    int tile = i >> 9;
    int nkb  = K >> 5;
    int kb   = tile % nkb;
    int nt   = tile / nkb;
    int row  = kb * 32 + (j & 8) * 2 + (lane >> 4) * 8 + (j & 7);
    int col  = nt * 16 + (lane & 15);
    Wp[i] = (__bf16)W[row * C + col];
}

__global__ void embed_kernel(const int* __restrict__ at_no, const float* __restrict__ emb,
                             float* __restrict__ x) {
    int i = blockIdx.x * blockDim.x + threadIdx.x;
    if (i < N_NODES * DIM) {
        int node = i >> 7, d = i & 127;
        x[i] = emb[at_no[node] * DIM + d];
    }
}

__global__ void edge_geom_kernel(const float* __restrict__ pos, const int* __restrict__ eidx,
                                 float* __restrict__ dirv, float* __restrict__ rbf,
                                 float* __restrict__ fcut) {
    int e = blockIdx.x * blockDim.x + threadIdx.x;
    if (e >= N_EDGES) return;
    int s = eidx[e];
    int d = eidx[N_EDGES + e];
    float vx = pos[d * 3 + 0] - pos[s * 3 + 0];
    float vy = pos[d * 3 + 1] - pos[s * 3 + 1];
    float vz = pos[d * 3 + 2] - pos[s * 3 + 2];
    float dist = sqrtf(vx * vx + vy * vy + vz * vz + 1e-12f);
    float inv = 1.0f / dist;
    dirv[e * 3 + 0] = vx * inv;
    dirv[e * 3 + 1] = vy * inv;
    dirv[e * 3 + 2] = vz * inv;
    float arg = PI_F * dist / CUTOFF;
#pragma unroll
    for (int n = 1; n <= NB; ++n)
        rbf[e * NB + (n - 1)] = __sinf((float)n * arg) * inv;
    fcut[e] = (dist < CUTOFF) ? 0.5f * (__cosf(arg) + 1.0f) : 0.0f;
}

// stage a 16xDIM f32 tile into LDS as bf16 using 128-bit loads
__device__ inline void stage_tile_bf16(const float* __restrict__ src, int row0, int ld,
                                       __bf16* dst, int t) {
    for (int q = t; q < 16 * DIM / 4; q += 128) {
        int m = (q * 4) >> 7, d = (q * 4) & 127;
        f4 val = *(const f4*)(src + (size_t)(row0 + m) * ld + d);
        v4bf w = { (__bf16)val.x, (__bf16)val.y, (__bf16)val.z, (__bf16)val.w };
        *(v4bf*)(dst + q * 4) = w;
    }
}

// ---------------------------------------------------------------------------
// phi = silu(x@Wm1 + bm1) @ Wm2 + bm2        (16-node tile per block, 4 waves)
// ---------------------------------------------------------------------------
__global__ __launch_bounds__(128)
void phi_kernel(const float* __restrict__ x, const __bf16* __restrict__ Wm1p,
                const float* __restrict__ bm1, const __bf16* __restrict__ Wm2p,
                const float* __restrict__ bm2, float* __restrict__ phi) {
    __shared__ __bf16 xs[16 * DIM];
    __shared__ __bf16 hs[16 * DIM];
    int row0 = blockIdx.x * 16;
    int t = threadIdx.x, lane = t & 31, wave = t >> 5;

    stage_tile_bf16(x, row0, DIM, xs, t);
    __syncthreads();

    for (int nt = wave; nt < 8; nt += 4) {            // 16x128 output
        v8f acc = {0.f, 0.f, 0.f, 0.f, 0.f, 0.f, 0.f, 0.f};
#pragma unroll
        for (int kb = 0; kb < 4; ++kb) {
            v16bf a = a_frag(xs, DIM, kb, lane);
            v16bf b = b_frag_packed(Wm1p, 4, kb, nt, lane);
            acc = WMMA_BF16(a, b, acc);
        }
        int n = (lane & 15) + nt * 16, half = lane >> 4;
        float bias = bm1[n];
#pragma unroll
        for (int r = 0; r < 8; ++r)
            hs[(r + 8 * half) * DIM + n] = (__bf16)silu_f(acc[r] + bias);
    }
    __syncthreads();

    for (int nt = wave; nt < 24; nt += 4) {           // 16x384 output
        v8f acc = {0.f, 0.f, 0.f, 0.f, 0.f, 0.f, 0.f, 0.f};
#pragma unroll
        for (int kb = 0; kb < 4; ++kb) {
            v16bf a = a_frag(hs, DIM, kb, lane);
            v16bf b = b_frag_packed(Wm2p, 4, kb, nt, lane);
            acc = WMMA_BF16(a, b, acc);
        }
        int n = (lane & 15) + nt * 16, half = lane >> 4;
        float bias = bm2[n];
#pragma unroll
        for (int r = 0; r < 8; ++r)
            phi[(size_t)(row0 + r + 8 * half) * D3 + n] = acc[r] + bias;
    }
}

// ---------------------------------------------------------------------------
// Fused edge message: filt = (rbf@Wf + bf)*fcut ; m = phi[src]*filt ;
// scatter ds -> xmsg[dst], dv -> vmsg[dst].  128 threads, 16 edges per block.
// Wf block (30KB) staged with CDNA5 async global->LDS DMA.
// ---------------------------------------------------------------------------
__global__ __launch_bounds__(128)
void message_kernel(const int* __restrict__ eidx, const float* __restrict__ phi,
                    const float* __restrict__ v, const float* __restrict__ dirv,
                    const float* __restrict__ rbf, const float* __restrict__ fcut,
                    const float* __restrict__ Wf, const float* __restrict__ bfv,
                    float* __restrict__ xmsg, float* __restrict__ vmsg) {
    __shared__ float wf_s[NB * D3];        // 30 KB
    __shared__ float rbf_s[EPB][NB];
    __shared__ float fc_s[EPB];
    __shared__ float dir_s[EPB][3];
    __shared__ int   src_s[EPB], dst_s[EPB];

    int t = threadIdx.x;
    int e0 = blockIdx.x * EPB;

    // async DMA: 30720 B = 1920 x 16B chunks, 15 per thread
    {
        uint32_t lds_base = (uint32_t)(uintptr_t)&wf_s[0];
#pragma unroll
        for (int q = 0; q < (NB * D3) / (4 * 128); ++q) {
            int chunk = q * 128 + t;
            async_copy_b128_to_lds(lds_base + chunk * 16, Wf + chunk * 4);
        }
    }
    for (int i = t; i < EPB * NB; i += 128) {
        int j = i / NB, k = i % NB;
        rbf_s[j][k] = rbf[(e0 + j) * NB + k];
    }
    if (t < EPB) {
        int e = e0 + t;
        fc_s[t] = fcut[e];
        dir_s[t][0] = dirv[e * 3 + 0];
        dir_s[t][1] = dirv[e * 3 + 1];
        dir_s[t][2] = dirv[e * 3 + 2];
        src_s[t] = eidx[e];
        dst_s[t] = eidx[N_EDGES + e];
    }
    wait_asynccnt0();
    __syncthreads();

    float b0 = bfv[t], b1 = bfv[DIM + t], b2 = bfv[2 * DIM + t];

    for (int j = 0; j < EPB; ++j) {
        int s = src_s[j], d = dst_s[j];
        if (j + 1 < EPB) {   // prefetch next edge's phi row (global_prefetch_b8)
            __builtin_prefetch(phi + (size_t)src_s[j + 1] * D3 + t, 0, 3);
        }
        float fc = fc_s[j];
        float f0 = b0, f1 = b1, f2 = b2;
#pragma unroll
        for (int k = 0; k < NB; ++k) {
            float r = rbf_s[j][k];
            f0 = fmaf(r, wf_s[k * D3 + t], f0);
            f1 = fmaf(r, wf_s[k * D3 + DIM + t], f1);
            f2 = fmaf(r, wf_s[k * D3 + 2 * DIM + t], f2);
        }
        f0 *= fc; f1 *= fc; f2 *= fc;
        const float* ps = phi + (size_t)s * D3;
        float ds  = ps[t] * f0;
        float dv1 = ps[DIM + t] * f1;
        float dv2 = ps[2 * DIM + t] * f2;
        atomic_add_f32(&xmsg[(size_t)d * DIM + t], ds);
#pragma unroll
        for (int c = 0; c < 3; ++c) {
            float dv = fmaf(dv2, v[((size_t)s * 3 + c) * DIM + t], dv1 * dir_s[j][c]);
            atomic_add_f32(&vmsg[((size_t)d * 3 + c) * DIM + t], dv);
        }
    }
}

// ---------------------------------------------------------------------------
// Node update: apply messages, Uv=v@U, Vv=v@V, a=silu([x,Vn]@Wu1+bu1)@Wu2+bu2,
// x += ass + asv*sum_c(Uv*Vv), v += avv*Uv.   16-node tile, 4 waves.
// ---------------------------------------------------------------------------
__global__ __launch_bounds__(128)
void update_kernel(float* __restrict__ x, float* __restrict__ v,
                   const float* __restrict__ xmsg, const float* __restrict__ vmsg,
                   const __bf16* __restrict__ Up, const __bf16* __restrict__ Vp,
                   const __bf16* __restrict__ Wu1p, const float* __restrict__ bu1,
                   const __bf16* __restrict__ Wu2p, const float* __restrict__ bu2) {
    __shared__ __bf16 vbuf[3 * 16 * DIM];  // 12 KB; reused as a2(16x256)+h(16x128)
    __shared__ __bf16 uvb[3 * 16 * DIM];   // 12 KB
    __shared__ __bf16 vvb[3 * 16 * DIM];   // 12 KB
    __shared__ float  abuf[16 * D3];       // 24 KB
    __bf16* a2 = vbuf;                      // 16*256 elems
    __bf16* hb = vbuf + 16 * 256;           // 16*128 elems

    int row0 = blockIdx.x * 16;
    int t = threadIdx.x, lane = t & 31, wave = t >> 5;

    // Phase A: v += vmsg (write back), stage bf16 v tile; 128-bit traffic
    for (int q = t; q < (3 * 16 * DIM) / 4; q += 128) {
        int rowv = q >> 5;               // (c,m) row, 48 rows
        int c = rowv >> 4, m = rowv & 15;
        int d = (q & 31) * 4;
        size_t g = (((size_t)(row0 + m) * 3 + c) * DIM + d);
        f4 val = *(const f4*)(v + g);
        f4 msg = *(const f4*)(vmsg + g);
        val.x += msg.x; val.y += msg.y; val.z += msg.z; val.w += msg.w;
        *(f4*)(v + g) = val;
        v4bf w = { (__bf16)val.x, (__bf16)val.y, (__bf16)val.z, (__bf16)val.w };
        *(v4bf*)(vbuf + q * 4) = w;
    }
    __syncthreads();

    // Phase B: Uv / Vv GEMMs (48 output tiles across 4 waves)
    for (int job = wave; job < 48; job += 4) {
        int which = job & 1;
        int c  = (job >> 1) / 8;
        int nt = (job >> 1) & 7;
        const __bf16* B = which ? Vp : Up;
        v8f acc = {0.f, 0.f, 0.f, 0.f, 0.f, 0.f, 0.f, 0.f};
#pragma unroll
        for (int kb = 0; kb < 4; ++kb) {
            v16bf a = a_frag(vbuf + c * 2048, DIM, kb, lane);
            v16bf b = b_frag_packed(B, 4, kb, nt, lane);
            acc = WMMA_BF16(a, b, acc);
        }
        __bf16* dst = (which ? vvb : uvb) + c * 2048;
        int n = (lane & 15) + nt * 16, half = lane >> 4;
#pragma unroll
        for (int r = 0; r < 8; ++r)
            dst[(r + 8 * half) * DIM + n] = (__bf16)acc[r];
    }
    __syncthreads();

    // Phase C: x += xmsg (write back); stage [x, Vn] into a2 (reuses vbuf)
    for (int q = t; q < (16 * DIM) / 4; q += 128) {
        int i = q * 4;
        int m = i >> 7, d = i & 127;
        size_t g = (size_t)(row0 + m) * DIM + d;
        f4 xv = *(const f4*)(x + g);
        f4 xm = *(const f4*)(xmsg + g);
        xv.x += xm.x; xv.y += xm.y; xv.z += xm.z; xv.w += xm.w;
        *(f4*)(x + g) = xv;
        f4 s = {1e-8f, 1e-8f, 1e-8f, 1e-8f};
#pragma unroll
        for (int c = 0; c < 3; ++c) {
            v4bf qv = *(const v4bf*)(vvb + c * 2048 + i);
            float q0 = (float)qv[0], q1 = (float)qv[1], q2 = (float)qv[2], q3 = (float)qv[3];
            s.x = fmaf(q0, q0, s.x); s.y = fmaf(q1, q1, s.y);
            s.z = fmaf(q2, q2, s.z); s.w = fmaf(q3, q3, s.w);
        }
        v4bf wx = { (__bf16)xv.x, (__bf16)xv.y, (__bf16)xv.z, (__bf16)xv.w };
        *(v4bf*)(a2 + m * 256 + d) = wx;
        v4bf wn = { (__bf16)sqrtf(s.x), (__bf16)sqrtf(s.y),
                    (__bf16)sqrtf(s.z), (__bf16)sqrtf(s.w) };
        *(v4bf*)(a2 + m * 256 + DIM + d) = wn;
    }
    __syncthreads();

    // Phase D: h = silu(a2 @ Wu1 + bu1)   (K=256)
    for (int nt = wave; nt < 8; nt += 4) {
        v8f acc = {0.f, 0.f, 0.f, 0.f, 0.f, 0.f, 0.f, 0.f};
#pragma unroll
        for (int kb = 0; kb < 8; ++kb) {
            v16bf a = a_frag(a2, 256, kb, lane);
            v16bf b = b_frag_packed(Wu1p, 8, kb, nt, lane);
            acc = WMMA_BF16(a, b, acc);
        }
        int n = (lane & 15) + nt * 16, half = lane >> 4;
        float bias = bu1[n];
#pragma unroll
        for (int r = 0; r < 8; ++r)
            hb[(r + 8 * half) * DIM + n] = (__bf16)silu_f(acc[r] + bias);
    }
    __syncthreads();

    // Phase E: a = h @ Wu2 + bu2   (16x384)
    for (int nt = wave; nt < 24; nt += 4) {
        v8f acc = {0.f, 0.f, 0.f, 0.f, 0.f, 0.f, 0.f, 0.f};
#pragma unroll
        for (int kb = 0; kb < 4; ++kb) {
            v16bf a = a_frag(hb, DIM, kb, lane);
            v16bf b = b_frag_packed(Wu2p, 4, kb, nt, lane);
            acc = WMMA_BF16(a, b, acc);
        }
        int n = (lane & 15) + nt * 16, half = lane >> 4;
        float bias = bu2[n];
#pragma unroll
        for (int r = 0; r < 8; ++r)
            abuf[(r + 8 * half) * D3 + n] = acc[r] + bias;
    }
    __syncthreads();

    // Phase F: final x / v update (128-bit traffic)
    for (int q = t; q < (16 * DIM) / 4; q += 128) {
        int i = q * 4;
        int m = i >> 7, d = i & 127;
        f4 uvv[3], dot = {0.f, 0.f, 0.f, 0.f};
#pragma unroll
        for (int c = 0; c < 3; ++c) {
            v4bf uq = *(const v4bf*)(uvb + c * 2048 + i);
            v4bf vq = *(const v4bf*)(vvb + c * 2048 + i);
            uvv[c].x = (float)uq[0]; uvv[c].y = (float)uq[1];
            uvv[c].z = (float)uq[2]; uvv[c].w = (float)uq[3];
            dot.x = fmaf(uvv[c].x, (float)vq[0], dot.x);
            dot.y = fmaf(uvv[c].y, (float)vq[1], dot.y);
            dot.z = fmaf(uvv[c].z, (float)vq[2], dot.z);
            dot.w = fmaf(uvv[c].w, (float)vq[3], dot.w);
        }
        f4 avv = *(const f4*)(abuf + m * D3 + d);
        f4 asv = *(const f4*)(abuf + m * D3 + DIM + d);
        f4 ass = *(const f4*)(abuf + m * D3 + 2 * DIM + d);
        size_t g = (size_t)(row0 + m) * DIM + d;
        f4 xv = *(const f4*)(x + g);
        xv.x += ass.x + asv.x * dot.x;
        xv.y += ass.y + asv.y * dot.y;
        xv.z += ass.z + asv.z * dot.z;
        xv.w += ass.w + asv.w * dot.w;
        *(f4*)(x + g) = xv;
#pragma unroll
        for (int c = 0; c < 3; ++c) {
            size_t gv = ((size_t)(row0 + m) * 3 + c) * DIM + d;
            f4 vv = *(const f4*)(v + gv);
            vv.x = fmaf(avv.x, uvv[c].x, vv.x);
            vv.y = fmaf(avv.y, uvv[c].y, vv.y);
            vv.z = fmaf(avv.z, uvv[c].z, vv.z);
            vv.w = fmaf(avv.w, uvv[c].w, vv.w);
            *(f4*)(v + gv) = vv;
        }
    }
}

// ---------------------------------------------------------------------------
// Output head: out[batch] += silu(x@Wo1+bo1) @ Wo2 + bo2
// ---------------------------------------------------------------------------
__global__ __launch_bounds__(128)
void output_kernel(const float* __restrict__ x, const int* __restrict__ batch,
                   const __bf16* __restrict__ Wo1p, const float* __restrict__ bo1,
                   const float* __restrict__ Wo2, const float* __restrict__ bo2,
                   float* __restrict__ out) {
    __shared__ __bf16 xs[16 * DIM];
    __shared__ float hbuf[16 * HD];
    int row0 = blockIdx.x * 16;
    int t = threadIdx.x, lane = t & 31, wave = t >> 5;

    stage_tile_bf16(x, row0, DIM, xs, t);
    __syncthreads();

    {   // 4 column tiles of 16 over HD=64; one tile per wave
        int nt = wave;
        v8f acc = {0.f, 0.f, 0.f, 0.f, 0.f, 0.f, 0.f, 0.f};
#pragma unroll
        for (int kb = 0; kb < 4; ++kb) {
            v16bf a = a_frag(xs, DIM, kb, lane);
            v16bf b = b_frag_packed(Wo1p, 4, kb, nt, lane);
            acc = WMMA_BF16(a, b, acc);
        }
        int n = (lane & 15) + nt * 16, half = lane >> 4;
        float bias = bo1[n];
#pragma unroll
        for (int r = 0; r < 8; ++r)
            hbuf[(r + 8 * half) * HD + n] = silu_f(acc[r] + bias);
    }
    __syncthreads();

    if (t < 16) {
        int node = row0 + t;
        float s = bo2[0];
#pragma unroll
        for (int j = 0; j < HD; ++j) s = fmaf(hbuf[t * HD + j], Wo2[j], s);
        atomic_add_f32(&out[batch[node]], s);
    }
}

// ---------------------------------------------------------------------------
extern "C" void kernel_launch(void* const* d_in, const int* in_sizes, int n_in,
                              void* d_out, int out_size, void* d_ws, size_t ws_size,
                              hipStream_t stream) {
    const int*   at_no = (const int*)  d_in[0];
    const float* pos   = (const float*)d_in[1];
    const int*   eidx  = (const int*)  d_in[2];
    const int*   batch = (const int*)  d_in[3];
    const float* emb   = (const float*)d_in[4];
    const float* Wf    = (const float*)d_in[5];
    const float* bfv   = (const float*)d_in[6];
    const float* Wm1   = (const float*)d_in[7];
    const float* bm1   = (const float*)d_in[8];
    const float* Wm2   = (const float*)d_in[9];
    const float* bm2   = (const float*)d_in[10];
    const float* U     = (const float*)d_in[11];
    const float* V     = (const float*)d_in[12];
    const float* Wu1   = (const float*)d_in[13];
    const float* bu1   = (const float*)d_in[14];
    const float* Wu2   = (const float*)d_in[15];
    const float* bu2   = (const float*)d_in[16];
    const float* Wo1   = (const float*)d_in[17];
    const float* bo1   = (const float*)d_in[18];
    const float* Wo2   = (const float*)d_in[19];
    const float* bo2   = (const float*)d_in[20];

    char* ws = (char*)d_ws;
    size_t off = 0;
    auto alloc = [&](size_t bytes) -> void* {
        void* p = ws + off;
        off = (off + bytes + 255) & ~(size_t)255;
        return p;
    };
    float* x    = (float*)alloc((size_t)N_NODES * DIM * 4);
    float* v    = (float*)alloc((size_t)N_NODES * 3 * DIM * 4);
    float* phi  = (float*)alloc((size_t)N_NODES * D3 * 4);
    float* xmsg = (float*)alloc((size_t)N_NODES * DIM * 4);
    float* vmsg = (float*)alloc((size_t)N_NODES * 3 * DIM * 4);
    float* dirv = (float*)alloc((size_t)N_EDGES * 3 * 4);
    float* rbf  = (float*)alloc((size_t)N_EDGES * NB * 4);
    float* fcut = (float*)alloc((size_t)N_EDGES * 4);
    __bf16* Wm1p = (__bf16*)alloc((size_t)NLAYER * DIM * DIM * 2);
    __bf16* Wm2p = (__bf16*)alloc((size_t)NLAYER * DIM * D3 * 2);
    __bf16* Up   = (__bf16*)alloc((size_t)NLAYER * DIM * DIM * 2);
    __bf16* Vp   = (__bf16*)alloc((size_t)NLAYER * DIM * DIM * 2);
    __bf16* Wu1p = (__bf16*)alloc((size_t)NLAYER * 2 * DIM * DIM * 2);
    __bf16* Wu2p = (__bf16*)alloc((size_t)NLAYER * DIM * D3 * 2);
    __bf16* Wo1p = (__bf16*)alloc((size_t)DIM * HD * 2);

    auto pack = [&](const float* s, __bf16* d, int K, int C) {
        pack_b_kernel<<<(K * C + 255) / 256, 256, 0, stream>>>(s, d, K, C);
    };
    for (int l = 0; l < NLAYER; ++l) {
        pack(Wm1 + (size_t)l * DIM * DIM,     Wm1p + (size_t)l * DIM * DIM,     DIM,     DIM);
        pack(Wm2 + (size_t)l * DIM * D3,      Wm2p + (size_t)l * DIM * D3,      DIM,     D3);
        pack(U   + (size_t)l * DIM * DIM,     Up   + (size_t)l * DIM * DIM,     DIM,     DIM);
        pack(V   + (size_t)l * DIM * DIM,     Vp   + (size_t)l * DIM * DIM,     DIM,     DIM);
        pack(Wu1 + (size_t)l * 2 * DIM * DIM, Wu1p + (size_t)l * 2 * DIM * DIM, 2 * DIM, DIM);
        pack(Wu2 + (size_t)l * DIM * D3,      Wu2p + (size_t)l * DIM * D3,      DIM,     D3);
    }
    pack(Wo1, Wo1p, DIM, HD);

    hipMemsetAsync(v, 0, (size_t)N_NODES * 3 * DIM * 4, stream);
    hipMemsetAsync(d_out, 0, (size_t)NGRAPH * 4, stream);

    embed_kernel<<<(N_NODES * DIM + 255) / 256, 256, 0, stream>>>(at_no, emb, x);
    edge_geom_kernel<<<(N_EDGES + 255) / 256, 256, 0, stream>>>(pos, eidx, dirv, rbf, fcut);

    for (int l = 0; l < NLAYER; ++l) {
        hipMemsetAsync(xmsg, 0, (size_t)N_NODES * DIM * 4, stream);
        hipMemsetAsync(vmsg, 0, (size_t)N_NODES * 3 * DIM * 4, stream);

        phi_kernel<<<N_NODES / 16, 128, 0, stream>>>(
            x, Wm1p + (size_t)l * DIM * DIM, bm1 + (size_t)l * DIM,
            Wm2p + (size_t)l * DIM * D3, bm2 + (size_t)l * D3, phi);

        message_kernel<<<N_EDGES / EPB, 128, 0, stream>>>(
            eidx, phi, v, dirv, rbf, fcut,
            Wf + (size_t)l * NB * D3, bfv + (size_t)l * D3, xmsg, vmsg);

        update_kernel<<<N_NODES / 16, 128, 0, stream>>>(
            x, v, xmsg, vmsg,
            Up + (size_t)l * DIM * DIM, Vp + (size_t)l * DIM * DIM,
            Wu1p + (size_t)l * 2 * DIM * DIM, bu1 + (size_t)l * DIM,
            Wu2p + (size_t)l * DIM * D3, bu2 + (size_t)l * D3);
    }

    output_kernel<<<N_NODES / 16, 128, 0, stream>>>(
        x, batch, Wo1p, bo1, Wo2, bo2, (float*)d_out);
}
